// Attention_43490838839830
// MI455X (gfx1250) — compile-verified, matrix operand order checked by reading, not run on hardware
//
#include <hip/hip_runtime.h>

// Problem constants (match reference)
constexpr int BB = 8;
constexpr int LQ = 2048;
constexpr int LK = 2048;
constexpr int DD = 64;

typedef _Float16 v16h __attribute__((ext_vector_type(16)));
typedef float    v8f  __attribute__((ext_vector_type(8)));
typedef int      vi4  __attribute__((ext_vector_type(4)));   // NT-capable int4
typedef float    vf4  __attribute__((ext_vector_type(4)));   // NT-capable float4

// Score-row stride in LDS (floats); 2056 = 2048 + 8 to skew banks, 16B-multiple
constexpr int STRIDE = 2056;
constexpr int SMEM_B = 32 * STRIDE * 4;   // 263,168 B < 320 KB/WGP

// Load two contiguous 8-half (16B) chunks into one WMMA 16-half operand.
// Per CDNA5 ISA 16-bit A layout: lane L (M = L&15, g = L>>4) holds
// halves h -> K = (h&7) + 8*g + 16*(h>>3)  =>  chunks at K = 8g and K = 16+8g.
__device__ inline v16h pack2(const _Float16* p0, const _Float16* p1) {
    union { uint4 u[2]; v16h h; } x;
    x.u[0] = *reinterpret_cast<const uint4*>(p0);
    x.u[1] = *reinterpret_cast<const uint4*>(p1);
    return x.h;
}

// ---------------------------------------------------------------------------
// Kernel 1: cosine-normalize Q,K -> f16 ; transpose V -> f16 Vt[b][d][k]
// grid (LK/64, B), block 256
// ---------------------------------------------------------------------------
__global__ void __launch_bounds__(256)
prep_kernel(const float* __restrict__ q, const float* __restrict__ k,
            const float* __restrict__ v,
            _Float16* __restrict__ qh, _Float16* __restrict__ kh,
            _Float16* __restrict__ vt) {
    const int b    = blockIdx.y;
    const int r0   = blockIdx.x * 64;      // 64 rows per block
    const int t    = threadIdx.x;
    const int rowl = t >> 2;               // 0..63
    const int quar = t & 3;                // 4 threads / row
    const int d0   = quar * 16;

    __shared__ _Float16 sVt[64][80];       // [d][k], 160B row stride (16B-mult)

    const size_t rowBase = ((size_t)b * LQ + (r0 + rowl)) * DD + d0;

    // normalize Q and K rows (f32 math, f16 output)
    for (int which = 0; which < 2; ++which) {
        const float* src = which ? k : q;
        _Float16*    dst = which ? kh : qh;
        const float4* p = reinterpret_cast<const float4*>(src + rowBase);
        float4 f[4];
        float ss = 0.f;
#pragma unroll
        for (int i = 0; i < 4; ++i) {
            f[i] = p[i];
            ss += f[i].x*f[i].x + f[i].y*f[i].y + f[i].z*f[i].z + f[i].w*f[i].w;
        }
        ss += __shfl_xor(ss, 1, 4);
        ss += __shfl_xor(ss, 2, 4);
        const float inv = 1.0f / (sqrtf(ss) + 1e-13f);
        union { _Float16 h[16]; uint4 u[2]; } o;
#pragma unroll
        for (int i = 0; i < 4; ++i) {
            o.h[4*i+0] = (_Float16)(f[i].x * inv);
            o.h[4*i+1] = (_Float16)(f[i].y * inv);
            o.h[4*i+2] = (_Float16)(f[i].z * inv);
            o.h[4*i+3] = (_Float16)(f[i].w * inv);
        }
        uint4* op = reinterpret_cast<uint4*>(dst + rowBase);
        op[0] = o.u[0]; op[1] = o.u[1];
    }

    // V tile -> f16, transpose through LDS
    {
        const float4* p = reinterpret_cast<const float4*>(v + rowBase);
#pragma unroll
        for (int i = 0; i < 4; ++i) {
            float4 f = p[i];
            sVt[d0 + 4*i + 0][rowl] = (_Float16)f.x;
            sVt[d0 + 4*i + 1][rowl] = (_Float16)f.y;
            sVt[d0 + 4*i + 2][rowl] = (_Float16)f.z;
            sVt[d0 + 4*i + 3][rowl] = (_Float16)f.w;
        }
    }
    __syncthreads();
    {
        const int d    = t >> 2;           // 0..63
        const int part = t & 3;            // 16 k-values each
        const uint4* s = reinterpret_cast<const uint4*>(&sVt[d][part * 16]);
        uint4* gdst = reinterpret_cast<uint4*>(
            vt + ((size_t)b * DD + d) * LK + r0 + part * 16);
        gdst[0] = s[0]; gdst[1] = s[1];
    }
}

// ---------------------------------------------------------------------------
// Kernel 2: fused attention for one (batch, 32-query tile)
// grid (LQ/32, B), block 256 (8 waves), dynamic LDS = SMEM_B
// ---------------------------------------------------------------------------
__global__ void __launch_bounds__(256, 1)
attn_kernel(const _Float16* __restrict__ qh, const _Float16* __restrict__ kh,
            const _Float16* __restrict__ vt, const int* __restrict__ mask,
            float* __restrict__ out, float* __restrict__ pat) {
    extern __shared__ float sS[];          // 32 x STRIDE scores
    const int b    = blockIdx.y;
    const int q0   = blockIdx.x * 32;
    const int t    = threadIdx.x;
    // wave id is uniform per wave -> force scalar so loops stay SALU-driven
    const int w    = __builtin_amdgcn_readfirstlane(t >> 5);   // 0..7
    const int lane = t & 31;
    const int ln   = lane & 15;
    const int g    = lane >> 4;
    const int r    = w & 1;                // row-tile owned by this wave

    // ---- Phase A: S = Qhat * Khat^T (raw scores into LDS) ----
    {
        const _Float16* qrow = qh + ((size_t)b * LQ + q0 + r*16 + ln) * DD;
        const v16h a0 = pack2(qrow + 8*g,      qrow + 16 + 8*g);   // d 0..31
        const v16h a1 = pack2(qrow + 32 + 8*g, qrow + 48 + 8*g);   // d 32..63
        for (int c = (w >> 1); c < 128; c += 4) {
            const _Float16* krow = kh + ((size_t)b * LK + c*16 + ln) * DD;
            v16h b0 = pack2(krow + 8*g,      krow + 16 + 8*g);
            v16h b1 = pack2(krow + 32 + 8*g, krow + 48 + 8*g);
            v8f acc = {};
            acc = __builtin_amdgcn_wmma_f32_16x16x32_f16(
                false, a0, false, b0, (short)0, acc, false, false);
            acc = __builtin_amdgcn_wmma_f32_16x16x32_f16(
                false, a1, false, b1, (short)0, acc, false, false);
            // C layout: lane -> N = ln, rows M = vv + 8*g
            float* dst = &sS[(r*16 + 8*g) * STRIDE + c*16 + ln];
#pragma unroll
            for (int vv = 0; vv < 8; ++vv) dst[vv * STRIDE] = acc[vv];
        }
    }
    __syncthreads();

    // ---- Phase B: mask + softmax; stream mask in / p_attn out (NT policy) ----
    {
        const int row = t >> 3;            // 0..31
        const int sub = t & 7;             // 8 lanes per row (one wave / 4 rows)
        float* srow = &sS[row * STRIDE];
        const int* mrow = mask + ((size_t)b * LQ + q0 + row) * LK;
        float*     prow = pat  + ((size_t)b * LQ + q0 + row) * LK;

        float m = -3.4e38f;
#pragma unroll 4
        for (int i = 0; i < 64; ++i) {
            const int col = (i * 8 + sub) * 4;
            if (col + 1024 < LK)           // stream-ahead prefetch of mask
                __builtin_prefetch(mrow + col + 1024, 0, 1);
            vi4 mk = __builtin_nontemporal_load(
                reinterpret_cast<const vi4*>(mrow + col));
            float4 s  = *reinterpret_cast<const float4*>(srow + col);
            s.x = mk.x ? s.x : -1e9f;
            s.y = mk.y ? s.y : -1e9f;
            s.z = mk.z ? s.z : -1e9f;
            s.w = mk.w ? s.w : -1e9f;
            *reinterpret_cast<float4*>(srow + col) = s;
            m = fmaxf(m, fmaxf(fmaxf(s.x, s.y), fmaxf(s.z, s.w)));
        }
        m = fmaxf(m, __shfl_xor(m, 1, 8));
        m = fmaxf(m, __shfl_xor(m, 2, 8));
        m = fmaxf(m, __shfl_xor(m, 4, 8));

        float sum = 0.f;
#pragma unroll 4
        for (int i = 0; i < 64; ++i) {
            const int col = (i * 8 + sub) * 4;
            float4 s = *reinterpret_cast<const float4*>(srow + col);
            s.x = __expf(s.x - m); s.y = __expf(s.y - m);
            s.z = __expf(s.z - m); s.w = __expf(s.w - m);
            *reinterpret_cast<float4*>(srow + col) = s;
            sum += s.x + s.y + s.z + s.w;
        }
        sum += __shfl_xor(sum, 1, 8);
        sum += __shfl_xor(sum, 2, 8);
        sum += __shfl_xor(sum, 4, 8);
        const float inv = 1.0f / sum;

        // Pass 3: normalize; emit p_attn (f32, NT) and compact P to packed f16
        // in-place at the head of the same LDS row. Safe within a lockstep
        // wave: f16 writes at iter i touch float slots <= 16i+15, f32 reads at
        // iter j>i start at slot 32j.
        _Float16* hrow = reinterpret_cast<_Float16*>(srow);
#pragma unroll 4
        for (int i = 0; i < 64; ++i) {
            const int col = (i * 8 + sub) * 4;
            float4 s = *reinterpret_cast<const float4*>(srow + col);
            s.x *= inv; s.y *= inv; s.z *= inv; s.w *= inv;
            vf4 sv;
            sv.x = s.x; sv.y = s.y; sv.z = s.z; sv.w = s.w;
            __builtin_nontemporal_store(
                sv, reinterpret_cast<vf4*>(prow + col));
            union { _Float16 h[4]; uint2 u; } p16;
            p16.h[0] = (_Float16)s.x; p16.h[1] = (_Float16)s.y;
            p16.h[2] = (_Float16)s.z; p16.h[3] = (_Float16)s.w;
            *reinterpret_cast<uint2*>(hrow + col) = p16.u;
        }
    }
    __syncthreads();

    // ---- Phase C: O = P * V ; one 16x16 tile per wave (r = w&1, c = w>>1) ----
    {
        const int c = w >> 1;              // 0..3 over D
        v8f acc = {};
        const _Float16* vrow = vt + ((size_t)b * DD + c*16 + ln) * LK;
        const _Float16* prow16 =
            reinterpret_cast<const _Float16*>(&sS[(r*16 + ln) * STRIDE]);
        for (int kk = 0; kk < LK; kk += 32) {
            v16h aa = pack2(prow16 + kk + 8*g, prow16 + kk + 16 + 8*g);
            v16h bb = pack2(vrow   + kk + 8*g, vrow   + kk + 16 + 8*g);
            acc = __builtin_amdgcn_wmma_f32_16x16x32_f16(
                false, aa, false, bb, (short)0, acc, false, false);
        }
        float* orow = out + ((size_t)b * LQ + q0 + r*16 + 8*g) * DD + c*16 + ln;
#pragma unroll
        for (int vv = 0; vv < 8; ++vv)
            __builtin_nontemporal_store(acc[vv], orow + (size_t)vv * DD);
    }
}

// ---------------------------------------------------------------------------
extern "C" void kernel_launch(void* const* d_in, const int* in_sizes, int n_in,
                              void* d_out, int out_size, void* d_ws, size_t ws_size,
                              hipStream_t stream) {
    const float* q    = (const float*)d_in[0];
    const float* k    = (const float*)d_in[1];
    const float* v    = (const float*)d_in[2];
    const int*   mask = (const int*)d_in[3];

    float* out = (float*)d_out;                       // (B,LQ,D)
    float* pat = out + (size_t)BB * LQ * DD;          // (B,LQ,LK)

    _Float16* qh = (_Float16*)d_ws;                   // 2 MB
    _Float16* kh = qh + (size_t)BB * LQ * DD;         // 2 MB
    _Float16* vt = kh + (size_t)BB * LK * DD;         // 2 MB (Vt[b][d][k])

    (void)hipFuncSetAttribute((const void*)attn_kernel,
                              hipFuncAttributeMaxDynamicSharedMemorySize,
                              SMEM_B);

    prep_kernel<<<dim3(LK / 64, BB), 256, 0, stream>>>(q, k, v, qh, kh, vt);
    attn_kernel<<<dim3(LQ / 32, BB), 256, SMEM_B, stream>>>(qh, kh, vt, mask,
                                                            out, pat);
}
